// EnhancedKalmanPredictorMambaBlockV0_36404142801628
// MI455X (gfx1250) — matrix operand order repaired and checked by reading, not compiled, not run on hardware
//
#include <hip/hip_runtime.h>
#include <math.h>

#define BB 2
#define HH 96
#define WW 96
#define LL (HH*WW)          // 9216
#define EE 96
#define DD 192
#define NNS 8
#define RRR 6
#define NLAYER 2
#define CPROJ (RRR + 2*NNS) // 22
#define CINC 4
#define MROWS (BB*LL)       // 18432

typedef __attribute__((ext_vector_type(16))) _Float16 v16h;
typedef __attribute__((ext_vector_type(8)))  _Float16 v8h;
typedef __attribute__((ext_vector_type(8)))  float    v8f;

__device__ __forceinline__ float lrelu_f(float v){ return v > 0.f ? v : 0.01f*v; }
__device__ __forceinline__ float silu_f(float v){ return v / (1.f + __expf(-v)); }
__device__ __forceinline__ float softplus_f(float v){ return v > 20.f ? v : log1pf(__expf(v)); }

// Spatial index read by (and written by) scan direction k at step l.
__device__ __forceinline__ int scan_map(int k, int l){
  int l2 = (k >= 2) ? (LL - 1 - l) : l;          // directions 2,3 are reversed
  if (k & 1) {                                    // directions 1,3 are transposed (w-major)
    int h = l2 % HH; int w = l2 / HH;
    return h*WW + w;
  }
  return l2;
}

// ---------------- f16 staging for WMMA ----------------
__global__ void k_cvt_f16(const float* __restrict__ in, _Float16* __restrict__ out, long n){
  long t = (long)blockIdx.x*blockDim.x + threadIdx.x;
  if (t >= n) return;
  out[t] = (_Float16)in[t];
}

// in [K,N] f32 row-major  ->  out [N,K] f16 ("column major" for fast K-runs)
__global__ void k_transpose_cvt(const float* __restrict__ in, _Float16* __restrict__ out,
                                int K, int N){
  long t = (long)blockIdx.x*blockDim.x + threadIdx.x;
  if (t >= (long)K*N) return;
  int n = (int)(t % N); int k = (int)(t / N);
  out[(size_t)n*K + k] = (_Float16)in[t];
}

// ---------------- WMMA GEMM: C[M,N] = A16[M,K] * Bt16[N,K]^T (+ optional addsrc*ss) -------
// One wave computes a 16 x (16*NT) strip of C. KTOT compile-time -> fully unrolled K loop.
template<int KTOT, int NT, bool HASADD>
__global__ void k_wmma_gemm_t(const _Float16* __restrict__ A16,
                              const _Float16* __restrict__ Bt16,
                              float* __restrict__ Cm, int M, int N,
                              const float* __restrict__ addsrc,
                              const float* __restrict__ ssp, int ssidx){
  long gid = (long)blockIdx.x*blockDim.x + threadIdx.x;
  int wave = (int)(gid >> 5);
  int lane = (int)(gid & 31);
  int ngroups = N / (16*NT);
  int mtiles  = M >> 4;
  if (wave >= ngroups*mtiles) return;
  int tm = wave / ngroups, tg = wave % ngroups;
  int khalf = lane >> 4;
  int l15   = lane & 15;
  // A layout (16-bit A 16x32): lane m = lane&15; per 32-chunk lane covers
  // K = khalf*8 + [0..7] and K = 16 + khalf*8 + [0..7]  -> two contiguous 8-half runs.
  const _Float16* arow = A16 + (size_t)(tm*16 + l15)*KTOT;
  v8f acc[NT];
  #pragma unroll
  for (int i = 0; i < NT; ++i) acc[i] = (v8f){};
  #pragma unroll
  for (int k0 = 0; k0 < KTOT; k0 += 32){
    v8h alo = *(const v8h*)(arow + k0 + khalf*8);
    v8h ahi = *(const v8h*)(arow + k0 + 16 + khalf*8);
    v16h af = __builtin_shufflevector(alo, ahi, 0,1,2,3,4,5,6,7,8,9,10,11,12,13,14,15);
    if (k0 + 32 < KTOT) __builtin_prefetch(arow + k0 + 32, 0, 1);
    #pragma unroll
    for (int nt = 0; nt < NT; ++nt){
      // B layout (16-bit B 32x16): lane n = lane&15; K = khalf*8 + [0..7] and 16 + khalf*8 + [0..7] (matches A).
      int ncol = (tg*NT + nt)*16 + l15;
      const _Float16* bcol = Bt16 + (size_t)ncol*KTOT + k0;
      v8h blo = *(const v8h*)(bcol + khalf*8);
      v8h bhi = *(const v8h*)(bcol + 16 + khalf*8);
      v16h bf = __builtin_shufflevector(blo, bhi, 0,1,2,3,4,5,6,7,8,9,10,11,12,13,14,15);
      acc[nt] = __builtin_amdgcn_wmma_f32_16x16x32_f16(false, af, false, bf,
                                                       (short)0, acc[nt], false, false);
    }
  }
  // C layout: VGPR r -> M = tm*16 + khalf*8 + r, N = ncol
  int mBase = tm*16 + khalf*8;
  #pragma unroll
  for (int nt = 0; nt < NT; ++nt){
    int ncol = (tg*NT + nt)*16 + l15;
    #pragma unroll
    for (int r2 = 0; r2 < 8; ++r2){
      size_t oi = (size_t)(mBase + r2)*N + ncol;
      float val = acc[nt][r2];
      if (HASADD) val += addsrc[oi]*ssp[ssidx];
      Cm[oi] = val;
    }
  }
}

// ---------------- encoder ----------------
__global__ void k_enc1(const float* __restrict__ img, const float* __restrict__ dz,
                       const float* __restrict__ w, const float* __restrict__ bias,
                       float* __restrict__ h1){
  long t = (long)blockIdx.x*blockDim.x + threadIdx.x;
  if (t >= (long)BB*EE*LL) return;
  int p = t % LL; int e = (t/LL) % EE; int b = t/((long)LL*EE);
  int h = p / WW, w_ = p % WW;
  float acc = bias[e];
  for (int kh = 0; kh < 3; ++kh){
    int hy = h + kh - 1; if (hy < 0 || hy >= HH) continue;
    for (int kw = 0; kw < 3; ++kw){
      int wx = w_ + kw - 1; if (wx < 0 || wx >= WW) continue;
      int sp = hy*WW + wx;
      const float* wp = w + ((e*2*CINC)*3 + kh)*3 + kw;   // [E,8,3,3]
      #pragma unroll
      for (int ci = 0; ci < CINC; ++ci)
        acc += img[(b*CINC+ci)*LL + sp] * wp[ci*9];
      #pragma unroll
      for (int ci = 0; ci < CINC; ++ci)
        acc += dz[(b*CINC+ci)*LL + sp] * wp[(CINC+ci)*9];
    }
  }
  h1[t] = lrelu_f(acc);
}

// h2 = conv3x3(h1) ; x = lrelu(h2 + conv1x1(x0)) written NHWC [B,L,E]
__global__ void k_enc2(const float* __restrict__ h1, const float* __restrict__ img,
                       const float* __restrict__ dz,
                       const float* __restrict__ w2, const float* __restrict__ b2,
                       const float* __restrict__ wsk, const float* __restrict__ bsk,
                       float* __restrict__ x){
  long t = (long)blockIdx.x*blockDim.x + threadIdx.x;
  if (t >= (long)BB*LL*EE) return;
  int e = t % EE; int p = (t/EE) % LL; int b = t/((long)EE*LL);
  int h = p / WW, w_ = p % WW;
  float acc = b2[e];
  for (int kh = 0; kh < 3; ++kh){
    int hy = h + kh - 1; if (hy < 0 || hy >= HH) continue;
    for (int kw = 0; kw < 3; ++kw){
      int wx = w_ + kw - 1; if (wx < 0 || wx >= WW) continue;
      int sp = hy*WW + wx;
      const float* wp = w2 + ((e*EE)*3 + kh)*3 + kw;      // [E,E,3,3]
      for (int ci = 0; ci < EE; ++ci)
        acc += h1[((long)(b*EE+ci))*LL + sp] * wp[ci*9];
    }
  }
  float sk = bsk[e];
  #pragma unroll
  for (int ci = 0; ci < CINC; ++ci) sk += img[(b*CINC+ci)*LL + p] * wsk[e*2*CINC + ci];
  #pragma unroll
  for (int ci = 0; ci < CINC; ++ci) sk += dz[(b*CINC+ci)*LL + p]  * wsk[e*2*CINC + CINC + ci];
  x[t] = lrelu_f(acc + sk);
}

// ---------------- layernorm over last dim ----------------
__global__ void k_layernorm(const float* __restrict__ in, const float* __restrict__ g,
                            const float* __restrict__ bta, float* __restrict__ out,
                            int rows, int width){
  int r = blockIdx.x*blockDim.x + threadIdx.x;
  if (r >= rows) return;
  const float* ip = in + (size_t)r*width;
  float m = 0.f;
  for (int i = 0; i < width; ++i) m += ip[i];
  m /= (float)width;
  float v = 0.f;
  for (int i = 0; i < width; ++i){ float d = ip[i]-m; v += d*d; }
  v /= (float)width;
  float inv = rsqrtf(v + 1e-5f);
  float* op = out + (size_t)r*width;
  for (int i = 0; i < width; ++i) op[i] = (ip[i]-m)*inv*g[i] + bta[i];
}

// ---------------- depthwise conv 3x3 + silu: xxz(NHWC, first D cols) -> xc [B,D,L] --------
__global__ void k_dwconv(const float* __restrict__ xxz, const float* __restrict__ w,
                         const float* __restrict__ bias, float* __restrict__ xc){
  long t = (long)blockIdx.x*blockDim.x + threadIdx.x;
  if (t >= (long)BB*DD*LL) return;
  int p = t % LL; int dd = (t/LL) % DD; int b = t/((long)LL*DD);
  int h = p / WW, w_ = p % WW;
  float acc = bias[dd];
  for (int kh = 0; kh < 3; ++kh){
    int hy = h + kh - 1; if (hy < 0 || hy >= HH) continue;
    for (int kw = 0; kw < 3; ++kw){
      int wx = w_ + kw - 1; if (wx < 0 || wx >= WW) continue;
      acc += xxz[((size_t)(b*LL + hy*WW + wx))*(2*DD) + dd] * w[dd*9 + kh*3 + kw];
    }
  }
  xc[t] = silu_f(acc);
}

// ---------------- x_dbl[b,k,c,l] = sum_d xs(b,k,d,l) * xproj[k,c,d] ----------------
__global__ void k_xdbl(const float* __restrict__ xc, const float* __restrict__ xproj,
                       float* __restrict__ xdbl){
  long t = (long)blockIdx.x*blockDim.x + threadIdx.x;
  if (t >= (long)BB*4*CPROJ*LL) return;
  int l = t % LL; int c = (t/LL) % CPROJ; int k = (t/((long)LL*CPROJ)) % 4;
  int b = t/((long)LL*CPROJ*4);
  int sp = scan_map(k, l);
  const float* wp = xproj + (size_t)(k*CPROJ + c)*DD;
  const float* xp = xc + (size_t)b*DD*LL + sp;
  float acc = 0.f;
  for (int dd = 0; dd < DD; ++dd) acc += xp[(size_t)dd*LL] * wp[dd];
  xdbl[t] = acc;
}

// ---------------- selective scan: one thread per (b,k,d); accumulates into yt[B,L,D] ------
__global__ void k_scan(const float* __restrict__ xdbl, const float* __restrict__ xc,
                       const float* __restrict__ dtw, const float* __restrict__ dtb,
                       const float* __restrict__ Alog, const float* __restrict__ Dsv,
                       float* __restrict__ yt){
  int t = blockIdx.x*blockDim.x + threadIdx.x;
  if (t >= BB*4*DD) return;
  int dd = t % DD; int k = (t/DD) % 4; int b = t/(DD*4);
  float a[NNS], wdt[RRR], hst[NNS];
  #pragma unroll
  for (int n = 0; n < NNS; ++n) a[n] = -__expf(Alog[(size_t)(k*DD+dd)*NNS + n]);
  #pragma unroll
  for (int r = 0; r < RRR; ++r) wdt[r] = dtw[(size_t)(k*DD+dd)*RRR + r];
  float bdt = dtb[k*DD+dd];
  float dsk = Dsv[k*DD+dd];
  #pragma unroll
  for (int n = 0; n < NNS; ++n) hst[n] = 0.f;
  const float* xd  = xdbl + (size_t)(b*4+k)*CPROJ*LL;
  const float* xcb = xc + (size_t)b*DD*LL + (size_t)dd*LL;
  for (int l = 0; l < LL; ++l){
    float dtr = bdt;
    #pragma unroll
    for (int r = 0; r < RRR; ++r) dtr += xd[(size_t)r*LL + l]*wdt[r];
    float dt = softplus_f(dtr);
    int sp = scan_map(k, l);
    float u  = xcb[sp];
    float du = dt*u;
    float y = 0.f;
    #pragma unroll
    for (int n = 0; n < NNS; ++n){
      float Bv = xd[(size_t)(RRR+n)*LL + l];
      float Cv = xd[(size_t)(RRR+NNS+n)*LL + l];
      hst[n] = hst[n]*__expf(dt*a[n]) + du*Bv;
      y += hst[n]*Cv;
    }
    atomicAdd(&yt[((size_t)b*LL + sp)*DD + dd], y + dsk*u);
  }
}

// ---------------- onorm (over D) * silu(z), in place on yt ----------------
__global__ void k_onorm_silu(float* __restrict__ yt, const float* __restrict__ xxz,
                             const float* __restrict__ g, const float* __restrict__ bta){
  int r = blockIdx.x*blockDim.x + threadIdx.x;
  if (r >= MROWS) return;
  float* ip = yt + (size_t)r*DD;
  float m = 0.f;
  for (int i = 0; i < DD; ++i) m += ip[i];
  m /= (float)DD;
  float v = 0.f;
  for (int i = 0; i < DD; ++i){ float d = ip[i]-m; v += d*d; }
  v /= (float)DD;
  float inv = rsqrtf(v + 1e-5f);
  const float* zp = xxz + (size_t)r*(2*DD) + DD;
  for (int i = 0; i < DD; ++i)
    ip[i] = ((ip[i]-m)*inv*g[i] + bta[i]) * silu_f(zp[i]);
}

// ---------------- CAB ----------------
__global__ void k_cab1(const float* __restrict__ xn, const float* __restrict__ w,
                       const float* __restrict__ bias, float* __restrict__ t1){
  long t = (long)blockIdx.x*blockDim.x + threadIdx.x;
  if (t >= (long)BB*(EE/2)*LL) return;
  int p = t % LL; int co = (t/LL) % (EE/2); int b = t/((long)LL*(EE/2));
  int h = p / WW, w_ = p % WW;
  float acc = bias[co];
  for (int kh = 0; kh < 3; ++kh){
    int hy = h + kh - 1; if (hy < 0 || hy >= HH) continue;
    for (int kw = 0; kw < 3; ++kw){
      int wx = w_ + kw - 1; if (wx < 0 || wx >= WW) continue;
      const float* xp = xn + ((size_t)(b*LL + hy*WW + wx))*EE;
      const float* wp = w + ((co*EE)*3 + kh)*3 + kw;
      for (int ci = 0; ci < EE; ++ci) acc += xp[ci]*wp[ci*9];
    }
  }
  t1[t] = lrelu_f(acc);
}

__global__ void k_cab2(const float* __restrict__ t1, const float* __restrict__ w,
                       const float* __restrict__ bias, float* __restrict__ t2){
  long t = (long)blockIdx.x*blockDim.x + threadIdx.x;
  if (t >= (long)BB*EE*LL) return;
  int p = t % LL; int co = (t/LL) % EE; int b = t/((long)LL*EE);
  int h = p / WW, w_ = p % WW;
  float acc = bias[co];
  for (int kh = 0; kh < 3; ++kh){
    int hy = h + kh - 1; if (hy < 0 || hy >= HH) continue;
    for (int kw = 0; kw < 3; ++kw){
      int wx = w_ + kw - 1; if (wx < 0 || wx >= WW) continue;
      int sp = hy*WW + wx;
      const float* wp = w + ((co*(EE/2))*3 + kh)*3 + kw;
      for (int ci = 0; ci < EE/2; ++ci)
        acc += t1[((long)(b*(EE/2)+ci))*LL + sp] * wp[ci*9];
    }
  }
  t2[t] = acc;
}

__global__ void k_pool(const float* __restrict__ t2, float* __restrict__ pooled){
  __shared__ float sm[256];
  int bc = blockIdx.x;               // (b*EE + co)
  float s = 0.f;
  for (int p = threadIdx.x; p < LL; p += blockDim.x) s += t2[(size_t)bc*LL + p];
  sm[threadIdx.x] = s; __syncthreads();
  for (int off = 128; off > 0; off >>= 1){
    if ((int)threadIdx.x < off) sm[threadIdx.x] += sm[threadIdx.x + off];
    __syncthreads();
  }
  if (threadIdx.x == 0) pooled[bc] = sm[0] / (float)LL;
}

__global__ void k_catt(const float* __restrict__ pooled, const float* __restrict__ w1,
                       const float* __restrict__ b1, const float* __restrict__ w2,
                       const float* __restrict__ b2, float* __restrict__ att){
  int t = blockIdx.x*blockDim.x + threadIdx.x;
  if (t >= BB*EE) return;
  int co = t % EE; int b = t / EE;
  float a1 = b1[0];
  for (int c = 0; c < EE; ++c) a1 += pooled[b*EE + c]*w1[c];
  a1 = fmaxf(a1, 0.f);
  float v = w2[co]*a1 + b2[co];
  att[t] = 1.f/(1.f + __expf(-v));
}

__global__ void k_cab_combine(float* __restrict__ x, const float* __restrict__ t2,
                              const float* __restrict__ att, const float* __restrict__ ssp,
                              int ssidx){
  long t = (long)blockIdx.x*blockDim.x + threadIdx.x;
  if (t >= (long)MROWS*EE) return;
  int e = t % EE; long m = t / EE; int b = (int)(m / LL); int p = (int)(m % LL);
  x[t] = x[t]*ssp[ssidx] + t2[((size_t)(b*EE + e))*LL + p]*att[b*EE + e];
}

// ---------------- head conv + tanh ----------------
__global__ void k_head(const float* __restrict__ x, const float* __restrict__ w,
                       const float* __restrict__ bias, float* __restrict__ out){
  long t = (long)blockIdx.x*blockDim.x + threadIdx.x;
  if (t >= (long)BB*CINC*LL) return;
  int p = t % LL; int co = (t/LL) % CINC; int b = t/((long)LL*CINC);
  int h = p / WW, w_ = p % WW;
  float acc = bias[co];
  for (int kh = 0; kh < 3; ++kh){
    int hy = h + kh - 1; if (hy < 0 || hy >= HH) continue;
    for (int kw = 0; kw < 3; ++kw){
      int wx = w_ + kw - 1; if (wx < 0 || wx >= WW) continue;
      const float* xp = x + ((size_t)(b*LL + hy*WW + wx))*EE;
      const float* wp = w + ((co*EE)*3 + kh)*3 + kw;
      for (int e = 0; e < EE; ++e) acc += xp[e]*wp[e*9];
    }
  }
  out[t] = tanhf(acc);
}

extern "C" void kernel_launch(void* const* d_in, const int* in_sizes, int n_in,
                              void* d_out, int out_size, void* d_ws, size_t ws_size,
                              hipStream_t stream) {
  (void)in_sizes; (void)n_in; (void)out_size; (void)ws_size;
  const float* image   = (const float*)d_in[0];
  const float* dz      = (const float*)d_in[1];
  const float* ec1_w   = (const float*)d_in[2];
  const float* ec1_b   = (const float*)d_in[3];
  const float* ec2_w   = (const float*)d_in[4];
  const float* ec2_b   = (const float*)d_in[5];
  const float* esk_w   = (const float*)d_in[6];
  const float* esk_b   = (const float*)d_in[7];
  const float* sh_w    = (const float*)d_in[8];
  const float* sh_b    = (const float*)d_in[9];
  const float* ln1_g   = (const float*)d_in[10];
  const float* ln1_b   = (const float*)d_in[11];
  const float* in_w    = (const float*)d_in[12];
  const float* dw_w    = (const float*)d_in[13];
  const float* dw_b    = (const float*)d_in[14];
  const float* xproj_w = (const float*)d_in[15];
  const float* dt_w    = (const float*)d_in[16];
  const float* dt_b    = (const float*)d_in[17];
  const float* A_log   = (const float*)d_in[18];
  const float* Ds      = (const float*)d_in[19];
  const float* onorm_g = (const float*)d_in[20];
  const float* onorm_b = (const float*)d_in[21];
  const float* out_w   = (const float*)d_in[22];
  const float* ss1     = (const float*)d_in[23];
  const float* ss2     = (const float*)d_in[24];
  const float* ln2_g   = (const float*)d_in[25];
  const float* ln2_b   = (const float*)d_in[26];
  const float* cab_w1  = (const float*)d_in[27];
  const float* cab_b1  = (const float*)d_in[28];
  const float* cab_w2  = (const float*)d_in[29];
  const float* cab_b2  = (const float*)d_in[30];
  const float* ca_w1   = (const float*)d_in[31];
  const float* ca_b1   = (const float*)d_in[32];
  const float* ca_w2   = (const float*)d_in[33];
  const float* ca_b2   = (const float*)d_in[34];

  float* ws = (float*)d_ws;
  size_t off = 0;
  auto alloc = [&](size_t n){ float* p = ws + off; off += (n + 3) & ~(size_t)3; return p; };
  float* x      = alloc((size_t)MROWS*EE);        // NHWC state
  float* xn     = alloc((size_t)MROWS*EE);        // layernorm output
  float* xxz    = alloc((size_t)MROWS*2*DD);      // in-proj output [xx | z]
  float* xc     = alloc((size_t)BB*DD*LL);        // dw-conv+silu, [B,D,L]
  float* xdbl   = alloc((size_t)BB*4*CPROJ*LL);   // [B,4,22,L]
  float* yt     = alloc((size_t)MROWS*DD);        // combined scan output [B,L,D]
  float* t2     = alloc((size_t)BB*EE*LL);        // also encoder h1
  float* t1     = alloc((size_t)BB*(EE/2)*LL);
  float* pooled = alloc((size_t)BB*EE);
  float* att    = alloc((size_t)BB*EE);
  _Float16* a16  = (_Float16*)alloc((size_t)MROWS*DD/2);   // f16 A staging (max 18432x192)
  _Float16* bt16 = (_Float16*)alloc((size_t)EE*2*DD/2);    // f16 B^T staging (max 96x384)

  const int TPB = 256;
  auto grd = [](long n, int tpb){ return (unsigned)((n + tpb - 1)/tpb); };

  // ---- encoder ----
  k_enc1<<<grd((long)BB*EE*LL,TPB),TPB,0,stream>>>(image, dz, ec1_w, ec1_b, t2);
  k_enc2<<<grd((long)BB*LL*EE,TPB),TPB,0,stream>>>(t2, image, dz, ec2_w, ec2_b, esk_w, esk_b, x);

  for (int i = 0; i < NLAYER; ++i){
    // ---- ln1 -> in-proj GEMM (WMMA f16): xn[M,96] @ in_w[96,384] -> xxz ----
    k_layernorm<<<grd(MROWS,TPB),TPB,0,stream>>>(x, ln1_g + i*EE, ln1_b + i*EE, xn, MROWS, EE);
    k_cvt_f16<<<grd((long)MROWS*EE,TPB),TPB,0,stream>>>(xn, a16, (long)MROWS*EE);
    k_transpose_cvt<<<grd((long)EE*2*DD,TPB),TPB,0,stream>>>(in_w + (size_t)i*EE*2*DD, bt16,
        EE, 2*DD);
    {
      long waves = (long)(MROWS/16) * ((2*DD)/64);   // NT=4
      k_wmma_gemm_t<EE, 4, false><<<grd(waves*32,TPB),TPB,0,stream>>>(
          a16, bt16, xxz, MROWS, 2*DD, nullptr, nullptr, 0);
    }
    k_dwconv<<<grd((long)BB*DD*LL,TPB),TPB,0,stream>>>(xxz, dw_w + (size_t)i*DD*9, dw_b + i*DD, xc);
    k_xdbl<<<grd((long)BB*4*CPROJ*LL,TPB),TPB,0,stream>>>(xc, xproj_w + (size_t)i*4*CPROJ*DD, xdbl);
    hipMemsetAsync(yt, 0, (size_t)MROWS*DD*sizeof(float), stream);
    k_scan<<<grd((long)BB*4*DD,TPB),TPB,0,stream>>>(xdbl, xc,
        dt_w + (size_t)i*4*DD*RRR, dt_b + (size_t)i*4*DD,
        A_log + (size_t)i*4*DD*NNS, Ds + (size_t)i*4*DD, yt);
    k_onorm_silu<<<grd(MROWS,TPB),TPB,0,stream>>>(yt, xxz, onorm_g + i*DD, onorm_b + i*DD);
    // ---- out-proj GEMM (WMMA f16) fused residual: x = x*ss1[i] + yt[M,192] @ out_w[192,96]
    k_cvt_f16<<<grd((long)MROWS*DD,TPB),TPB,0,stream>>>(yt, a16, (long)MROWS*DD);
    k_transpose_cvt<<<grd((long)DD*EE,TPB),TPB,0,stream>>>(out_w + (size_t)i*DD*EE, bt16,
        DD, EE);
    {
      long waves = (long)(MROWS/16) * (EE/32);       // NT=2
      k_wmma_gemm_t<DD, 2, true><<<grd(waves*32,TPB),TPB,0,stream>>>(
          a16, bt16, x, MROWS, EE, x, ss1, i);
    }
    // ---- ln2 -> CAB ----
    k_layernorm<<<grd(MROWS,TPB),TPB,0,stream>>>(x, ln2_g + i*EE, ln2_b + i*EE, xn, MROWS, EE);
    k_cab1<<<grd((long)BB*(EE/2)*LL,TPB),TPB,0,stream>>>(xn, cab_w1 + (size_t)i*(EE/2)*EE*9,
        cab_b1 + i*(EE/2), t1);
    k_cab2<<<grd((long)BB*EE*LL,TPB),TPB,0,stream>>>(t1, cab_w2 + (size_t)i*EE*(EE/2)*9,
        cab_b2 + i*EE, t2);
    k_pool<<<BB*EE,256,0,stream>>>(t2, pooled);
    k_catt<<<grd((long)BB*EE,TPB),TPB,0,stream>>>(pooled, ca_w1 + (size_t)i*EE, ca_b1 + i,
        ca_w2 + (size_t)i*EE, ca_b2 + i*EE, att);
    k_cab_combine<<<grd((long)MROWS*EE,TPB),TPB,0,stream>>>(x, t2, att, ss2, i);
  }

  // ---- head ----
  k_head<<<grd((long)BB*CINC*LL,TPB),TPB,0,stream>>>(x, sh_w, sh_b, (float*)d_out);
}
